// view_transformer_34961033789535
// MI455X (gfx1250) — matrix operand order repaired and checked by reading, not compiled, not run on hardware
//
#include <hip/hip_runtime.h>
#include <stdint.h>

// ---------------- problem constants ----------------
static constexpr int Bb   = 16;
static constexpr int Nn   = 1024;
static constexpr int DM   = 512;
static constexpr int DK   = 256;
static constexpr int DV   = 256;
static constexpr int Hh   = 4;
static constexpr int DFF  = 2048;
static constexpr int KNNK = 20;
static constexpr int ROWS = Bb * Nn;      // 16384
static constexpr float LNEPS = 1e-5f;

typedef __attribute__((ext_vector_type(16))) __bf16 v16bf;
typedef __attribute__((ext_vector_type(8)))  float  v8f;

union Frag16 {                 // bf16 A/B operand: 16 bf16 per lane = 8 VGPRs
    v16bf v;
    uint4 q[2];
    unsigned short u[16];
};
union FragAcc {                // f32 C/D: 8 f32 per lane = 8 VGPRs
    v8f   v;
    float f[8];
};

__device__ __forceinline__ unsigned short f2bf(float x) {
    uint32_t u = __float_as_uint(x);
    u += 0x7FFFu + ((u >> 16) & 1u);     // round-to-nearest-even
    return (unsigned short)(u >> 16);
}

__device__ __forceinline__ v8f wmma_bf16(const Frag16& a, const Frag16& b, const v8f& c) {
    return __builtin_amdgcn_wmma_f32_16x16x32_bf16(false, a.v, false, b.v,
                                                   (short)0, c, false, false);
}

// ---- CDNA5 async global->LDS (bypasses VGPRs, tracked by ASYNCcnt) ----
// LDS address operand = low 32 bits of the generic pointer (LDS aperture truncation).
__device__ __forceinline__ void async_b128(const void* gsrc, const void* lds) {
    asm volatile("global_load_async_to_lds_b128 %0, %1, off"
                 :: "v"((uint32_t)(uintptr_t)lds), "v"(gsrc)
                 : "memory");
}
__device__ __forceinline__ void wait_async0() {
    asm volatile("s_wait_asynccnt 0x0" ::: "memory");
}

// ---------------- cast fp32 -> bf16 ----------------
__global__ void cast_bf16_kernel(const float* __restrict__ src,
                                 unsigned short* __restrict__ dst, int n) {
    int i = (blockIdx.x * blockDim.x + threadIdx.x) * 4;
    if (i + 3 < n) {
        float4 v = *(const float4*)(src + i);
        dst[i + 0] = f2bf(v.x); dst[i + 1] = f2bf(v.y);
        dst[i + 2] = f2bf(v.z); dst[i + 3] = f2bf(v.w);
    }
}

// ---------------- KNN mask: one wave per destination point ----------------
// maskw[(b*N+n)*32 + w] bit m%32 of word m/32 == 1  =>  key m allowed for query n
__global__ __launch_bounds__(256) void knn_mask_kernel(const float* __restrict__ verts,
                                                       uint32_t* __restrict__ maskw) {
    const int wid  = threadIdx.x >> 5;
    const int lane = threadIdx.x & 31;
    const int p    = blockIdx.x * 8 + wid;        // b*N + n
    const int b    = p >> 10;
    const int n    = p & 1023;
    const float* vb = verts + (size_t)b * Nn * 3;
    const float cx = vb[n * 3 + 0], cy = vb[n * 3 + 1], cz = vb[n * 3 + 2];

    float d[32];
#pragma unroll
    for (int i = 0; i < 32; ++i) {
        int m = i * 32 + lane;
        float dx = vb[m * 3 + 0] - cx;
        float dy = vb[m * 3 + 1] - cy;
        float dz = vb[m * 3 + 2] - cz;
        d[i] = dx * dx + dy * dy + dz * dz;
    }
    uint32_t word = 0;
    for (int it = 0; it < KNNK; ++it) {
        float bd = __builtin_inff();
        int   bi = 0x7fffffff;
#pragma unroll
        for (int i = 0; i < 32; ++i) {
            int m = i * 32 + lane;
            if (d[i] < bd || (d[i] == bd && m < bi)) { bd = d[i]; bi = m; }
        }
#pragma unroll
        for (int off = 16; off; off >>= 1) {
            float od = __shfl_xor(bd, off, 32);
            int   oi = __shfl_xor(bi, off, 32);
            if (od < bd || (od == bd && oi < bi)) { bd = od; bi = oi; }
        }
        if ((bi & 31) == lane) d[bi >> 5] = __builtin_inff();   // remove winner
        if ((bi >> 5) == lane) word |= 1u << (bi & 31);         // set mask bit
    }
    maskw[(size_t)p * 32 + lane] = word;
}

// ---------------- bf16 GEMM: C[M,Nc] = A[M,Kc] * B[Kc,Nc] + bias ----------------
// Double-buffered LDS; A staged by async global->LDS, B (transposing gather) pipelined
// through VGPRs. MODE 0: f32 out  1: bf16 out  2: bf16+relu  3: bf16 transposed (V->Vt)
template <int MODE>
__global__ __launch_bounds__(256) void gemm_bf16_kernel(
        const unsigned short* __restrict__ A, const unsigned short* __restrict__ Bw,
        const float* __restrict__ bias, float* __restrict__ Cf,
        unsigned short* __restrict__ Cb, int M, int Ncols, int Kc) {
    __shared__ unsigned short Alds[2][128 * 32];    // [row][k]
    __shared__ unsigned short Blds[2][128 * 32];    // BT: [n][k]

    const int tid  = threadIdx.x;
    const int wid  = tid >> 5;
    const int lane = tid & 31;
    const int lo   = lane & 15, hi = lane >> 4;
    const int wm   = wid & 3;                    // 4 waves along M
    const int wn   = wid >> 2;                   // 2 waves along N
    const int bm   = blockIdx.y * 128;
    const int bn   = blockIdx.x * 128;

    FragAcc acc[2][4];
#pragma unroll
    for (int ms = 0; ms < 2; ++ms)
#pragma unroll
        for (int ns = 0; ns < 4; ++ns)
#pragma unroll
            for (int r = 0; r < 8; ++r) acc[ms][ns].f[r] = 0.0f;

    const int arow = tid >> 1, ahalf = tid & 1;  // A: 2 threads/row, 16 ushorts each
    const int btn  = tid & 127, bkg = tid >> 7;  // B: col slice, 16 k's each

    union Pack { unsigned short u[16]; uint4 q[2]; };

    // ---- prologue: stage tile kb=0 into buffer 0 ----
    {
        const unsigned short* src = A + (size_t)(bm + arow) * Kc + ahalf * 16;
        async_b128(src + 0, &Alds[0][arow * 32 + ahalf * 16 + 0]);
        async_b128(src + 8, &Alds[0][arow * 32 + ahalf * 16 + 8]);
        Pack pk;
#pragma unroll
        for (int i = 0; i < 16; ++i)
            pk.u[i] = Bw[(size_t)(bkg * 16 + i) * Ncols + bn + btn];
        *(uint4*)&Blds[0][btn * 32 + bkg * 16 + 0] = pk.q[0];
        *(uint4*)&Blds[0][btn * 32 + bkg * 16 + 8] = pk.q[1];
    }
    wait_async0();
    __syncthreads();

    const int nk = Kc >> 5;
    for (int ik = 0; ik < nk; ++ik) {
        const int cur = ik & 1, nxt = cur ^ 1;
        const int kb  = ik << 5;
        const bool hasNext = (ik + 1 < nk);

        Pack pk;
        if (hasNext) {
            // issue next A tile async + next B tile global loads before compute
            const unsigned short* src = A + (size_t)(bm + arow) * Kc + kb + 32 + ahalf * 16;
            async_b128(src + 0, &Alds[nxt][arow * 32 + ahalf * 16 + 0]);
            async_b128(src + 8, &Alds[nxt][arow * 32 + ahalf * 16 + 8]);
#pragma unroll
            for (int i = 0; i < 16; ++i)
                pk.u[i] = Bw[(size_t)(kb + 32 + bkg * 16 + i) * Ncols + bn + btn];
        }

        Frag16 bf[4];
#pragma unroll
        for (int ns = 0; ns < 4; ++ns) {
            int col = wn * 64 + ns * 16 + lo;
            bf[ns].q[0] = *(const uint4*)&Blds[cur][col * 32 + hi * 8];
            bf[ns].q[1] = *(const uint4*)&Blds[cur][col * 32 + 16 + hi * 8];
        }
#pragma unroll
        for (int ms = 0; ms < 2; ++ms) {
            Frag16 af;
            int row = wm * 32 + ms * 16 + lo;
            af.q[0] = *(const uint4*)&Alds[cur][row * 32 + hi * 8];
            af.q[1] = *(const uint4*)&Alds[cur][row * 32 + 16 + hi * 8];
#pragma unroll
            for (int ns = 0; ns < 4; ++ns)
                acc[ms][ns].v = wmma_bf16(af, bf[ns], acc[ms][ns].v);
        }

        if (hasNext) {
            *(uint4*)&Blds[nxt][btn * 32 + bkg * 16 + 0] = pk.q[0];
            *(uint4*)&Blds[nxt][btn * 32 + bkg * 16 + 8] = pk.q[1];
        }
        wait_async0();          // next A tile arrived (overlapped with WMMAs above)
        __syncthreads();
    }

#pragma unroll
    for (int ms = 0; ms < 2; ++ms)
#pragma unroll
        for (int ns = 0; ns < 4; ++ns)
#pragma unroll
            for (int r = 0; r < 8; ++r) {
                int gRow = bm + wm * 32 + ms * 16 + hi * 8 + r;
                int gCol = bn + wn * 64 + ns * 16 + lo;
                float val = acc[ms][ns].f[r] + bias[gCol];
                if (MODE == 0) {
                    Cf[(size_t)gRow * Ncols + gCol] = val;
                } else if (MODE == 1) {
                    Cb[(size_t)gRow * Ncols + gCol] = f2bf(val);
                } else if (MODE == 2) {
                    Cb[(size_t)gRow * Ncols + gCol] = f2bf(fmaxf(val, 0.0f));
                } else {                                  // V -> Vt[B,H,DV,N]
                    int h = gCol >> 8, dv = gCol & 255;
                    int b = gRow >> 10, n = gRow & 1023;
                    Cb[((size_t)((b * Hh + h) * DV + dv) << 10) + n] = f2bf(val);
                }
            }
}

// ---------------- fused flash attention ----------------
// grid: (N/128, H, B); 8 waves x 16 query rows; key tiles of 32; K/Vt staged by
// async global->LDS with double buffering.
__global__ __launch_bounds__(256) void attn_kernel(
        const unsigned short* __restrict__ Q, const unsigned short* __restrict__ Kmat,
        const unsigned short* __restrict__ Vt, const uint32_t* __restrict__ maskw,
        unsigned short* __restrict__ Obf) {
    __shared__ unsigned short Klds[2][32 * 256];   // [keyLocal][dk]   2 x 16KB
    __shared__ unsigned short Vlds[2][256 * 32];   // [dv][keyLocal]   2 x 16KB
    __shared__ unsigned short Plds[8][16 * 32];    // per-wave P scratch, 8KB

    const int tid  = threadIdx.x;
    const int wid  = tid >> 5;
    const int lane = tid & 31;
    const int lo   = lane & 15, hi = lane >> 4;
    const int b    = blockIdx.z, h = blockIdx.y;
    const int q0   = blockIdx.x * 128 + wid * 16;
    const size_t qrowbase = (size_t)b * Nn + q0;
    const float scale = 0.0625f;                  // 1/sqrt(256)

    // async-stage one 32-key tile (K: 32x256, Vt: 256x32) into buffer `buf`
    auto stage = [&](int buf, int kt) {
        int key = tid >> 3, part = tid & 7;
        const unsigned short* ksrc =
            Kmat + ((size_t)b * Nn + kt + key) * (Hh * DK) + h * DK + part * 32;
#pragma unroll
        for (int c = 0; c < 4; ++c)
            async_b128(ksrc + c * 8, &Klds[buf][key * 256 + part * 32 + c * 8]);
        int dv = tid;
        const unsigned short* vsrc = Vt + ((size_t)((b * Hh + h) * DV + dv) << 10) + kt;
#pragma unroll
        for (int c = 0; c < 4; ++c)
            async_b128(vsrc + c * 8, &Vlds[buf][dv * 32 + c * 8]);
    };

    // Q fragments: 16 rows x 256 dk, 8 fragments of 16x32
    Frag16 qf[8];
#pragma unroll
    for (int t = 0; t < 8; ++t) {
        const unsigned short* qp = Q + (qrowbase + lo) * (Hh * DK) + h * DK + t * 32;
        qf[t].q[0] = *(const uint4*)(qp + hi * 8);
        qf[t].q[1] = *(const uint4*)(qp + 16 + hi * 8);
    }

    FragAcc o[16];
#pragma unroll
    for (int f = 0; f < 16; ++f)
#pragma unroll
        for (int r = 0; r < 8; ++r) o[f].f[r] = 0.0f;
    float mrun[8], lrun[8];
#pragma unroll
    for (int r = 0; r < 8; ++r) { mrun[r] = -__builtin_inff(); lrun[r] = 0.0f; }

    stage(0, 0);
    wait_async0();
    __syncthreads();

    for (int kt = 0; kt < Nn; kt += 32) {
        const int cur = (kt >> 5) & 1;
        if (kt + 32 < Nn) stage(cur ^ 1, kt + 32);   // overlap next tile with compute

        uint32_t mw[8];
#pragma unroll
        for (int r = 0; r < 8; ++r)
            mw[r] = maskw[(qrowbase + hi * 8 + r) * 32 + (kt >> 5)];

        // S = Q * K^T  (16 x 32)
        FragAcc s[2];
#pragma unroll
        for (int sub = 0; sub < 2; ++sub) {
#pragma unroll
            for (int r = 0; r < 8; ++r) s[sub].f[r] = 0.0f;
#pragma unroll
            for (int t = 0; t < 8; ++t) {
                Frag16 kf;
                int key = sub * 16 + lo;
                kf.q[0] = *(const uint4*)&Klds[cur][key * 256 + t * 32 + hi * 8];
                kf.q[1] = *(const uint4*)&Klds[cur][key * 256 + t * 32 + 16 + hi * 8];
                s[sub].v = wmma_bf16(qf[t], kf, s[sub].v);
            }
        }

        // masked online softmax (rows grouped per half-wave in C layout)
        float corrv[8];
#pragma unroll
        for (int r = 0; r < 8; ++r) {
            float s0 = s[0].f[r] * scale;
            float s1 = s[1].f[r] * scale;
            if (!((mw[r] >> lo) & 1u))        s0 = -__builtin_inff();
            if (!((mw[r] >> (16 + lo)) & 1u)) s1 = -__builtin_inff();
            float rmax = fmaxf(s0, s1);
#pragma unroll
            for (int off = 8; off; off >>= 1)
                rmax = fmaxf(rmax, __shfl_xor(rmax, off, 32));
            float mnew = fmaxf(mrun[r], rmax);
            float corr = (mnew == -__builtin_inff()) ? 1.0f : __expf(mrun[r] - mnew);
            float p0 = (s0 == -__builtin_inff()) ? 0.0f : __expf(s0 - mnew);
            float p1 = (s1 == -__builtin_inff()) ? 0.0f : __expf(s1 - mnew);
            float rs = p0 + p1;
#pragma unroll
            for (int off = 8; off; off >>= 1) rs += __shfl_xor(rs, off, 32);
            lrun[r] = lrun[r] * corr + rs;
            mrun[r] = mnew;
            corrv[r] = corr;
            int prow = hi * 8 + r;
            Plds[wid][prow * 32 + lo]      = f2bf(p0);
            Plds[wid][prow * 32 + 16 + lo] = f2bf(p1);
        }
#pragma unroll
        for (int f = 0; f < 16; ++f)
#pragma unroll
            for (int r = 0; r < 8; ++r) o[f].f[r] *= corrv[r];

        // reload P in A-fragment layout (same-wave LDS: in-order)
        Frag16 pf;
        pf.q[0] = *(const uint4*)&Plds[wid][lo * 32 + hi * 8];
        pf.q[1] = *(const uint4*)&Plds[wid][lo * 32 + 16 + hi * 8];

        // O += P * V   (16 x 256)
#pragma unroll
        for (int f = 0; f < 16; ++f) {
            Frag16 vf;
            int dv = f * 16 + lo;
            vf.q[0] = *(const uint4*)&Vlds[cur][dv * 32 + hi * 8];
            vf.q[1] = *(const uint4*)&Vlds[cur][dv * 32 + 16 + hi * 8];
            o[f].v = wmma_bf16(pf, vf, o[f].v);
        }

        wait_async0();           // next K/Vt tile arrived (overlapped)
        __syncthreads();
    }

    float inv[8];
#pragma unroll
    for (int r = 0; r < 8; ++r) inv[r] = (lrun[r] > 0.0f) ? 1.0f / lrun[r] : 0.0f;
#pragma unroll
    for (int f = 0; f < 16; ++f)
#pragma unroll
        for (int r = 0; r < 8; ++r) {
            size_t row = qrowbase + hi * 8 + r;
            int col = h * DV + f * 16 + lo;
            Obf[row * (Hh * DV) + col] = f2bf(o[f].f[r] * inv[r]);
        }
}

// ---------------- residual + layernorm: y = LN(res + a)*g + b ----------------
__global__ __launch_bounds__(256) void ln_kernel(
        const float* __restrict__ a, const float* __restrict__ res,
        const float* __restrict__ g, const float* __restrict__ be,
        float* __restrict__ y, unsigned short* __restrict__ ybf) {
    const int wid  = threadIdx.x >> 5;
    const int lane = threadIdx.x & 31;
    const size_t row = (size_t)blockIdx.x * 8 + wid;
    const size_t base = row * DM;

    float v[16];
    float s = 0.0f;
#pragma unroll
    for (int i = 0; i < 16; ++i) {
        int idx = lane + i * 32;
        v[i] = a[base + idx] + res[base + idx];
        s += v[i];
    }
#pragma unroll
    for (int off = 16; off; off >>= 1) s += __shfl_xor(s, off, 32);
    float mean = s * (1.0f / DM);
    float var = 0.0f;
#pragma unroll
    for (int i = 0; i < 16; ++i) { float d = v[i] - mean; var += d * d; }
#pragma unroll
    for (int off = 16; off; off >>= 1) var += __shfl_xor(var, off, 32);
    float rstd = rsqrtf(var * (1.0f / DM) + LNEPS);
#pragma unroll
    for (int i = 0; i < 16; ++i) {
        int idx = lane + i * 32;
        float out = (v[i] - mean) * rstd * g[idx] + be[idx];
        y[base + idx] = out;
        if (ybf) ybf[base + idx] = f2bf(out);
    }
}

// ---------------- launch ----------------
extern "C" void kernel_launch(void* const* d_in, const int* in_sizes, int n_in,
                              void* d_out, int out_size, void* d_ws, size_t ws_size,
                              hipStream_t stream) {
    (void)in_sizes; (void)n_in; (void)out_size; (void)ws_size;
    const float* vertices = (const float*)d_in[0];
    const float* F        = (const float*)d_in[1];
    const float* Wq = (const float*)d_in[2];  const float* bq = (const float*)d_in[3];
    const float* Wk = (const float*)d_in[4];  const float* bk = (const float*)d_in[5];
    const float* Wv = (const float*)d_in[6];  const float* bv = (const float*)d_in[7];
    const float* Wo = (const float*)d_in[8];  const float* bo = (const float*)d_in[9];
    const float* g1 = (const float*)d_in[10]; const float* be1 = (const float*)d_in[11];
    const float* W1 = (const float*)d_in[12]; const float* b1 = (const float*)d_in[13];
    const float* W2 = (const float*)d_in[14]; const float* b2 = (const float*)d_in[15];
    const float* g2 = (const float*)d_in[16]; const float* be2 = (const float*)d_in[17];
    float* out = (float*)d_out;

    char* wsp = (char*)d_ws;
    auto alloc = [&](size_t bytes) {
        void* r = (void*)wsp;
        wsp += (bytes + 255) & ~(size_t)255;
        return r;
    };
    uint32_t*       maskw   = (uint32_t*)alloc((size_t)ROWS * 32 * 4);
    unsigned short* Fbf     = (unsigned short*)alloc((size_t)ROWS * DM * 2);
    unsigned short* Wq_bf   = (unsigned short*)alloc((size_t)DM * Hh * DK * 2);
    unsigned short* Wk_bf   = (unsigned short*)alloc((size_t)DM * Hh * DK * 2);
    unsigned short* Wv_bf   = (unsigned short*)alloc((size_t)DM * Hh * DV * 2);
    unsigned short* Wo_bf   = (unsigned short*)alloc((size_t)Hh * DV * DM * 2);
    unsigned short* W1_bf   = (unsigned short*)alloc((size_t)DM * DFF * 2);
    unsigned short* W2_bf   = (unsigned short*)alloc((size_t)DFF * DM * 2);
    unsigned short* Qbf     = (unsigned short*)alloc((size_t)ROWS * Hh * DK * 2);
    unsigned short* Kbf     = (unsigned short*)alloc((size_t)ROWS * Hh * DK * 2);
    unsigned short* Vtbf    = (unsigned short*)alloc((size_t)ROWS * Hh * DV * 2);
    unsigned short* Abf     = (unsigned short*)alloc((size_t)ROWS * Hh * DV * 2);  // attn out
    float*          aproj   = (float*)alloc((size_t)ROWS * DM * 4);
    float*          xf      = (float*)alloc((size_t)ROWS * DM * 4);
    unsigned short* xbf     = (unsigned short*)alloc((size_t)ROWS * DM * 2);
    unsigned short* hbf     = (unsigned short*)alloc((size_t)ROWS * DFF * 2);
    float*          ff      = (float*)alloc((size_t)ROWS * DM * 4);

    auto cast = [&](const float* s, unsigned short* d, int n) {
        cast_bf16_kernel<<<(n / 4 + 255) / 256, 256, 0, stream>>>(s, d, n);
    };
    cast(F,  Fbf,   ROWS * DM);
    cast(Wq, Wq_bf, DM * Hh * DK);
    cast(Wk, Wk_bf, DM * Hh * DK);
    cast(Wv, Wv_bf, DM * Hh * DV);
    cast(Wo, Wo_bf, Hh * DV * DM);
    cast(W1, W1_bf, DM * DFF);
    cast(W2, W2_bf, DFF * DM);

    knn_mask_kernel<<<ROWS / 8, 256, 0, stream>>>(vertices, maskw);

    // Q, K, V projections (V written pre-transposed)
    gemm_bf16_kernel<1><<<dim3(Hh * DK / 128, ROWS / 128), 256, 0, stream>>>(
        Fbf, Wq_bf, bq, nullptr, Qbf, ROWS, Hh * DK, DM);
    gemm_bf16_kernel<1><<<dim3(Hh * DK / 128, ROWS / 128), 256, 0, stream>>>(
        Fbf, Wk_bf, bk, nullptr, Kbf, ROWS, Hh * DK, DM);
    gemm_bf16_kernel<3><<<dim3(Hh * DV / 128, ROWS / 128), 256, 0, stream>>>(
        Fbf, Wv_bf, bv, nullptr, Vtbf, ROWS, Hh * DV, DM);

    attn_kernel<<<dim3(Nn / 128, Hh, Bb), 256, 0, stream>>>(Qbf, Kbf, Vtbf, maskw, Abf);

    gemm_bf16_kernel<0><<<dim3(DM / 128, ROWS / 128), 256, 0, stream>>>(
        Abf, Wo_bf, bo, aproj, nullptr, ROWS, DM, Hh * DV);

    ln_kernel<<<ROWS / 8, 256, 0, stream>>>(aproj, F, g1, be1, xf, xbf);

    gemm_bf16_kernel<2><<<dim3(DFF / 128, ROWS / 128), 256, 0, stream>>>(
        xbf, W1_bf, b1, nullptr, hbf, ROWS, DFF, DM);
    gemm_bf16_kernel<0><<<dim3(DM / 128, ROWS / 128), 256, 0, stream>>>(
        hbf, W2_bf, b2, ff, nullptr, ROWS, DM, DFF);

    ln_kernel<<<ROWS / 8, 256, 0, stream>>>(ff, xf, g2, be2, out, nullptr);
}
